// CrossAttention_68126771249481
// MI455X (gfx1250) — matrix operand order repaired and checked by reading, not compiled, not run on hardware
//
#include <hip/hip_runtime.h>
#include <hip/hip_bf16.h>

// ---------------- problem constants ----------------
constexpr int Bb = 2;
constexpr int Tt = 2048;
constexpr int Cc = 1024;
constexpr int Hh = 16;
constexpr int Dd = 64;          // head dim
constexpr int BT = Bb * Tt;     // 4096 rows for projection GEMMs

typedef __attribute__((ext_vector_type(16))) _Float16 v16h;
typedef __attribute__((ext_vector_type(8)))  _Float16 v8h;
typedef __attribute__((ext_vector_type(8)))  float    v8f;

// ---------------- WMMA fragment loaders (wave32, 16x16x32 f16) ----------------
// A-matrix 16x32: lane<16 -> row=lane,   K = {0..7, 16..23}
//                 lane>=16 -> row=lane-16, K = {8..15, 24..31}
__device__ __forceinline__ v16h frag_a(const _Float16* rowBase, int lane) {
    const int kA = (lane >> 4) << 3;   // 0 or 8
    v8h lo = *reinterpret_cast<const v8h*>(rowBase + kA);
    v8h hi = *reinterpret_cast<const v8h*>(rowBase + kA + 16);
    return __builtin_shufflevector(lo, hi, 0,1,2,3,4,5,6,7,8,9,10,11,12,13,14,15);
}
// B-matrix 32x16 (stored transposed in LDS: colBase = &Bt[col][0]):
// lane<16 -> col=lane,    K = {0..7, 16..23} ; lane>=16 -> col=lane-16, K = {8..15, 24..31}
__device__ __forceinline__ v16h frag_b(const _Float16* colBase, int lane) {
    const int kB = (lane >> 4) << 3;   // 0 or 8
    v8h lo = *reinterpret_cast<const v8h*>(colBase + kB);
    v8h hi = *reinterpret_cast<const v8h*>(colBase + kB + 16);
    return __builtin_shufflevector(lo, hi, 0,1,2,3,4,5,6,7,8,9,10,11,12,13,14,15);
}

__device__ __forceinline__ v8f wmma_f16(v16h a, v16h b, v8f c) {
    return __builtin_amdgcn_wmma_f32_16x16x32_f16(false, a, false, b,
                                                  (short)0, c, false, false);
}

__device__ __forceinline__ v8f zero8() {
    v8f z; for (int i = 0; i < 8; ++i) z[i] = 0.0f; return z;
}

// ---------------- fp32 -> f16 conversion ----------------
__global__ void f32_to_f16(const float* __restrict__ src,
                           _Float16* __restrict__ dst, int n) {
    int i = blockIdx.x * blockDim.x + threadIdx.x;
    int stride = gridDim.x * blockDim.x;
    for (; i < n; i += stride) dst[i] = (_Float16)src[i];
}

// ---------------- tiled WMMA GEMM:  C[M,N] = A[M,K] @ B[K,N] + bias ----------------
// A row-major (lda=K), B row-major (ldb given), output row-major (ldc=N).
// Block tile 128x128, BK=32, 256 threads = 8 waves (4 along M x 2 along N),
// each wave computes 2x4 = 8 WMMA 16x16 tiles. Double-buffered LDS.
template <bool F32OUT>
__global__ __launch_bounds__(256) void gemm_wmma(
    const _Float16* __restrict__ A, const _Float16* __restrict__ Bm,
    const float* __restrict__ bias, void* __restrict__ Cout,
    int M, int N, int K, int ldb)
{
    constexpr int BM = 128, BN = 128, BK = 32;
    __shared__ alignas(32) _Float16 As[2][BM][BK];  // [row][k]
    __shared__ alignas(32) _Float16 Bs[2][BN][BK];  // transposed: [n][k]

    const int tid  = threadIdx.x;
    const int lane = tid & 31;
    const int w    = tid >> 5;
    const int wm   = (w & 3) * 32;   // wave row base within block tile
    const int wn   = (w >> 2) * 64;  // wave col base within block tile
    const int m0   = blockIdx.y * BM;
    const int n0   = blockIdx.x * BN;

    // A loader: 2 threads per row, 16 halves each
    const int arow = tid >> 1, aseg = (tid & 1) * 16;
    // B loader: 8 threads per k-row, 16 halves each, scattered (transpose) into LDS
    const int bk = tid >> 3, bn = (tid & 7) * 16;

    v8f acc[2][4];
    for (int mi = 0; mi < 2; ++mi)
        for (int ni = 0; ni < 4; ++ni) acc[mi][ni] = zero8();

    auto loadTile = [&](int kt, int buf) {
        const uint4* ga = reinterpret_cast<const uint4*>(
            A + (size_t)(m0 + arow) * K + kt * BK + aseg);
        uint4* la = reinterpret_cast<uint4*>(&As[buf][arow][aseg]);
        la[0] = ga[0]; la[1] = ga[1];

        _Float16 tmp[16];
        const uint4* gb = reinterpret_cast<const uint4*>(
            Bm + (size_t)(kt * BK + bk) * ldb + n0 + bn);
        reinterpret_cast<uint4*>(tmp)[0] = gb[0];
        reinterpret_cast<uint4*>(tmp)[1] = gb[1];
        #pragma unroll
        for (int j = 0; j < 16; ++j) Bs[buf][bn + j][bk] = tmp[j];
    };

    const int NT = K / BK;
    loadTile(0, 0);
    __syncthreads();

    for (int kt = 0; kt < NT; ++kt) {
        const int cur = kt & 1;
        if (kt + 1 < NT) loadTile(kt + 1, cur ^ 1);

        v16h af[2], bf[4];
        #pragma unroll
        for (int mi = 0; mi < 2; ++mi)
            af[mi] = frag_a(&As[cur][wm + mi * 16 + (lane & 15)][0], lane);
        #pragma unroll
        for (int ni = 0; ni < 4; ++ni)
            bf[ni] = frag_b(&Bs[cur][wn + ni * 16 + (lane & 15)][0], lane);

        #pragma unroll
        for (int mi = 0; mi < 2; ++mi)
            #pragma unroll
            for (int ni = 0; ni < 4; ++ni)
                acc[mi][ni] = wmma_f16(af[mi], bf[ni], acc[mi][ni]);

        __syncthreads();
    }

    // Epilogue. C/D layout: VGPR i -> row = (lane<16 ? i : i+8), col = lane&15
    const int hrow = (lane >> 4) * 8;
    #pragma unroll
    for (int mi = 0; mi < 2; ++mi) {
        #pragma unroll
        for (int ni = 0; ni < 4; ++ni) {
            #pragma unroll
            for (int i = 0; i < 8; ++i) {
                const int row = m0 + wm + mi * 16 + hrow + i;
                const int col = n0 + wn + ni * 16 + (lane & 15);
                const float v = acc[mi][ni][i] + bias[col];
                if constexpr (F32OUT)
                    reinterpret_cast<float*>(Cout)[(size_t)row * N + col] = v;
                else
                    reinterpret_cast<_Float16*>(Cout)[(size_t)row * N + col] = (_Float16)v;
            }
        }
    }
}

// ---------------- flash attention (per (b,h), Br=Bc=64, online softmax) ----------------
// 128 threads = 4 waves; wave w owns q-rows [w*16, w*16+16).
__global__ __launch_bounds__(128) void flash_attn(
    const _Float16* __restrict__ Q, const _Float16* __restrict__ Kmat,
    const _Float16* __restrict__ Vmat, _Float16* __restrict__ Y)
{
    constexpr int Br = 64, Bc = 64;
    __shared__ alignas(32) _Float16 Qs [Br][Dd];   // [qrow][d]
    __shared__ alignas(32) _Float16 Ks [Bc][Dd];   // [key][d]  (== K^T B-fragments)
    __shared__ alignas(32) _Float16 Vts[Dd][Bc];   // transposed: [d][key]
    __shared__ alignas(32) _Float16 Ps [4][16][Bc];// per-wave P staging (A-layout source)

    const int tid  = threadIdx.x;
    const int lane = tid & 31;
    const int w    = tid >> 5;
    const int b    = blockIdx.y / Hh;
    const int h    = blockIdx.y % Hh;
    const int q0   = blockIdx.x * Br;
    const size_t headOff = (size_t)b * Tt * Cc + (size_t)h * Dd;

    // load Q block (64 rows x 64 halves; 2 threads/row, 32 halves each)
    {
        const int r = tid >> 1, seg = (tid & 1) * 32;
        const uint4* g = reinterpret_cast<const uint4*>(
            Q + headOff + (size_t)(q0 + r) * Cc + seg);
        uint4* l = reinterpret_cast<uint4*>(&Qs[r][seg]);
        l[0] = g[0]; l[1] = g[1]; l[2] = g[2]; l[3] = g[3];
    }

    const int hrow = (lane >> 4) * 8;   // C-layout row offset for this lane half
    float mrow[8], lrow[8];
    v8f oacc[4];
    #pragma unroll
    for (int i = 0; i < 8; ++i) { mrow[i] = -1e30f; lrow[i] = 0.0f; }
    #pragma unroll
    for (int ni = 0; ni < 4; ++ni) oacc[ni] = zero8();

    for (int kb = 0; kb < Tt / Bc; ++kb) {
        __syncthreads();   // previous iteration's P*V reads done before overwrite
        {
            const int r = tid >> 1, seg = (tid & 1) * 32;
            const uint4* gk = reinterpret_cast<const uint4*>(
                Kmat + headOff + (size_t)(kb * Bc + r) * Cc + seg);
            uint4* lk = reinterpret_cast<uint4*>(&Ks[r][seg]);
            lk[0] = gk[0]; lk[1] = gk[1]; lk[2] = gk[2]; lk[3] = gk[3];

            _Float16 tmp[32];
            const uint4* gv = reinterpret_cast<const uint4*>(
                Vmat + headOff + (size_t)(kb * Bc + r) * Cc + seg);
            reinterpret_cast<uint4*>(tmp)[0] = gv[0];
            reinterpret_cast<uint4*>(tmp)[1] = gv[1];
            reinterpret_cast<uint4*>(tmp)[2] = gv[2];
            reinterpret_cast<uint4*>(tmp)[3] = gv[3];
            #pragma unroll
            for (int j = 0; j < 32; ++j) Vts[seg + j][r] = tmp[j];
        }
        __syncthreads();

        // S = (Q K^T) * 1/sqrt(D) : 16x64 per wave
        v8f sacc[4];
        #pragma unroll
        for (int ni = 0; ni < 4; ++ni) sacc[ni] = zero8();
        #pragma unroll
        for (int ds = 0; ds < Dd; ds += 32) {
            v16h aq = frag_a(&Qs[w * 16 + (lane & 15)][ds], lane);
            #pragma unroll
            for (int ni = 0; ni < 4; ++ni) {
                v16h bkf = frag_b(&Ks[ni * 16 + (lane & 15)][ds], lane);
                sacc[ni] = wmma_f16(aq, bkf, sacc[ni]);
            }
        }
        #pragma unroll
        for (int ni = 0; ni < 4; ++ni)
            #pragma unroll
            for (int i = 0; i < 8; ++i) sacc[ni][i] *= 0.125f;  // 1/sqrt(64)

        // online softmax (rows live across 16-lane halves -> xor-reduce 1,2,4,8)
        float fac[8];
        #pragma unroll
        for (int i = 0; i < 8; ++i) {
            float mx = fmaxf(fmaxf(sacc[0][i], sacc[1][i]),
                             fmaxf(sacc[2][i], sacc[3][i]));
            for (int off = 1; off < 16; off <<= 1)
                mx = fmaxf(mx, __shfl_xor(mx, off, 32));
            const float mnew = fmaxf(mrow[i], mx);
            fac[i] = __expf(mrow[i] - mnew);
            mrow[i] = mnew;
            float s = 0.0f;
            #pragma unroll
            for (int ni = 0; ni < 4; ++ni) {
                const float p = __expf(sacc[ni][i] - mnew);
                s += p;
                Ps[w][hrow + i][ni * 16 + (lane & 15)] = (_Float16)p;
            }
            for (int off = 1; off < 16; off <<= 1)
                s += __shfl_xor(s, off, 32);
            lrow[i] = lrow[i] * fac[i] + s;
            #pragma unroll
            for (int ni = 0; ni < 4; ++ni) oacc[ni][i] *= fac[i];
        }
        __syncthreads();   // Ps visible (uniform barrier across all 4 waves)

        // O += P @ V
        #pragma unroll
        for (int ks = 0; ks < Bc; ks += 32) {
            v16h ap = frag_a(&Ps[w][lane & 15][ks], lane);
            #pragma unroll
            for (int ni = 0; ni < 4; ++ni) {
                v16h bv = frag_b(&Vts[ni * 16 + (lane & 15)][ks], lane);
                oacc[ni] = wmma_f16(ap, bv, oacc[ni]);
            }
        }
    }

    // normalize and store y (f16, (B,T,C) layout)
    #pragma unroll
    for (int i = 0; i < 8; ++i) {
        const float inv = 1.0f / lrow[i];
        const int row = q0 + w * 16 + hrow + i;
        #pragma unroll
        for (int ni = 0; ni < 4; ++ni)
            Y[headOff + (size_t)row * Cc + ni * 16 + (lane & 15)] =
                (_Float16)(oacc[ni][i] * inv);
    }
}

// ---------------- host launcher ----------------
extern "C" void kernel_launch(void* const* d_in, const int* in_sizes, int n_in,
                              void* d_out, int out_size, void* d_ws, size_t ws_size,
                              hipStream_t stream) {
    (void)in_sizes; (void)n_in; (void)out_size; (void)ws_size;
    const float* x_q  = (const float*)d_in[0];
    const float* x_kv = (const float*)d_in[1];
    const float* Wq   = (const float*)d_in[2];
    const float* bq   = (const float*)d_in[3];
    const float* Wkv  = (const float*)d_in[4];
    const float* bkv  = (const float*)d_in[5];
    const float* Wo   = (const float*)d_in[6];
    const float* bo   = (const float*)d_in[7];
    float* out = (float*)d_out;

    constexpr int NBTC = BT * Cc;          // 4,194,304
    _Float16* ws = (_Float16*)d_ws;
    size_t o = 0;
    _Float16* xq_h  = ws + o; o += NBTC;
    _Float16* xkv_h = ws + o; o += NBTC;
    _Float16* Wq_h  = ws + o; o += Cc * Cc;
    _Float16* Wkv_h = ws + o; o += Cc * 2 * Cc;
    _Float16* Wo_h  = ws + o; o += Cc * Cc;
    _Float16* q_h   = ws + o; o += NBTC;
    _Float16* k_h   = ws + o; o += NBTC;
    _Float16* v_h   = ws + o; o += NBTC;
    _Float16* y_h   = ws + o; o += NBTC;

    f32_to_f16<<<1024, 256, 0, stream>>>(x_q,  xq_h,  NBTC);
    f32_to_f16<<<1024, 256, 0, stream>>>(x_kv, xkv_h, NBTC);
    f32_to_f16<<<512,  256, 0, stream>>>(Wq,   Wq_h,  Cc * Cc);
    f32_to_f16<<<512,  256, 0, stream>>>(Wkv,  Wkv_h, Cc * 2 * Cc);
    f32_to_f16<<<512,  256, 0, stream>>>(Wo,   Wo_h,  Cc * Cc);

    dim3 gproj(Cc / 128, BT / 128);   // (8, 32)
    gemm_wmma<false><<<gproj, 256, 0, stream>>>(xq_h,  Wq_h,       bq,      q_h, BT, Cc, Cc, Cc);
    gemm_wmma<false><<<gproj, 256, 0, stream>>>(xkv_h, Wkv_h,      bkv,     k_h, BT, Cc, Cc, 2 * Cc);
    gemm_wmma<false><<<gproj, 256, 0, stream>>>(xkv_h, Wkv_h + Cc, bkv + Cc, v_h, BT, Cc, Cc, 2 * Cc);

    flash_attn<<<dim3(Tt / 64, Bb * Hh), 128, 0, stream>>>(q_h, k_h, v_h, y_h);

    gemm_wmma<true><<<gproj, 256, 0, stream>>>(y_h, Wo_h, bo, out, BT, Cc, Cc, Cc);
}
